// MMSingleStreamBlock_82978768158932
// MI455X (gfx1250) — matrix operand order repaired and checked by reading, not compiled, not run on hardware
//
#include <hip/hip_runtime.h>

// ---------------------------------------------------------------------------
// Problem constants (from reference)
// ---------------------------------------------------------------------------
constexpr int LQ   = 2048;
constexpr int TXT  = 256;
constexpr int IMG  = LQ - TXT;       // 1792
constexpr int HID  = 3072;
constexpr int NH   = 24;
constexpr int HD   = 128;
constexpr int MLP  = 12288;
constexpr int NY   = 3 * HID + MLP;  // 21504 (y width: q|k|v|mlp)
constexpr int NC   = HID + MLP;      // 15360 (concat width: attn|gelu(mlp))
constexpr float EPSV = 1e-6f;

// Workspace layout (bytes). bf16 stored as unsigned short.
constexpr size_t OFF_MOD  = 0;                                   // 9216 f32
constexpr size_t OFF_XMOD = 65536;                               // L*HID bf16
constexpr size_t OFF_W1B  = OFF_XMOD + (size_t)LQ * HID * 2;     // HID*NY bf16
constexpr size_t OFF_Y    = OFF_W1B + (size_t)HID * NY * 2;      // L*NY bf16
constexpr size_t OFF_W2B  = OFF_Y + (size_t)LQ * NY * 2;         // NC*HID bf16
constexpr size_t OFF_CAT  = OFF_W2B + (size_t)NC * HID * 2;      // L*NC bf16

// ---------------------------------------------------------------------------
// Types / helpers
// ---------------------------------------------------------------------------
typedef __attribute__((ext_vector_type(16))) __bf16 v16bf;
typedef __attribute__((ext_vector_type(8)))  float  v8f;

union BF16Frag { uint4 u[2]; v16bf v; };

__device__ __forceinline__ unsigned short f2bf(float f) {
  unsigned int u = __float_as_uint(f);
  unsigned int r = (u + 0x7FFFu + ((u >> 16) & 1u)) >> 16;   // RNE
  return (unsigned short)r;
}
__device__ __forceinline__ float bf2f(unsigned short h) {
  return __uint_as_float(((unsigned int)h) << 16);
}

// ---------------------------------------------------------------------------
// 1) mod = silu(vec) @ w_mod + b_mod          (1x3072)x(3072x9216)
// ---------------------------------------------------------------------------
__global__ void k_mod(const float* __restrict__ vec, const float* __restrict__ w_mod,
                      const float* __restrict__ b_mod, float* __restrict__ mod) {
  __shared__ float sv[HID];
  int t = threadIdx.x;
  for (int i = t; i < HID; i += 256) {
    float v = vec[i];
    sv[i] = v / (1.f + __expf(-v));
  }
  __syncthreads();
  int j = blockIdx.x * 256 + t;                 // 0..9215
  float acc = b_mod[j];
  for (int i = 0; i < HID; ++i)
    acc = fmaf(sv[i], w_mod[(size_t)i * (3 * HID) + j], acc);
  mod[j] = acc;
}

// ---------------------------------------------------------------------------
// 2) f32 -> bf16 weight conversion (bf16 copy of w1/w2 fits in 192MB L2)
// ---------------------------------------------------------------------------
__global__ void k_cvt(const float* __restrict__ src, unsigned short* __restrict__ dst,
                      size_t n) {
  size_t i = ((size_t)blockIdx.x * 256 + threadIdx.x) * 4;
#pragma unroll
  for (int j = 0; j < 4; ++j)
    if (i + j < n) dst[i + j] = f2bf(src[i + j]);
}

// ---------------------------------------------------------------------------
// 3) LayerNorm + modulate -> x_mod (bf16). One block per row.
// ---------------------------------------------------------------------------
__global__ void k_ln(const float* __restrict__ x, const float* __restrict__ mod,
                     unsigned short* __restrict__ xmod) {
  int l = blockIdx.x, t = threadIdx.x;
  __shared__ float red[256];
  const float* row = x + (size_t)l * HID;
  float s = 0.f, s2 = 0.f;
  for (int i = t; i < HID; i += 256) { float v = row[i]; s += v; s2 += v * v; }
  red[t] = s; __syncthreads();
  for (int o = 128; o > 0; o >>= 1) { if (t < o) red[t] += red[t + o]; __syncthreads(); }
  float mean = red[0] / HID; __syncthreads();
  red[t] = s2; __syncthreads();
  for (int o = 128; o > 0; o >>= 1) { if (t < o) red[t] += red[t + o]; __syncthreads(); }
  float var = red[0] / HID - mean * mean;
  float rstd = rsqrtf(var + EPSV);
  const float* shift = mod;
  const float* scale = mod + HID;
  for (int i = t; i < HID; i += 256) {
    float v = (row[i] - mean) * rstd;
    xmod[(size_t)l * HID + i] = f2bf(v * (1.f + scale[i]) + shift[i]);
  }
}

// ---------------------------------------------------------------------------
// 4/8) WMMA GEMM: C[2048 x N] = A_bf16[2048 x K] @ B_bf16[K x N] + bias
//   block tile 128x128, 8 waves, wave tile 32x64, K-step 32.
//   EPI=0: store bf16 (y).  EPI=1: store f32 residual x + gate*val (d_out).
// ---------------------------------------------------------------------------
template <int EPI>
__global__ __launch_bounds__(256) void k_gemm(
    const unsigned short* __restrict__ A, const unsigned short* __restrict__ B,
    const float* __restrict__ bias, int K, int N,
    unsigned short* __restrict__ outbf,
    const float* __restrict__ xres, const float* __restrict__ gate,
    float* __restrict__ outf) {
  __shared__ unsigned short lA[128 * 32];     // [m][k]
  __shared__ unsigned short lB[128 * 32];     // [n][k] (transposed)
  int m0 = blockIdx.y * 128, n0 = blockIdx.x * 128;
  int t = threadIdx.x, lane = t & 31, w = t >> 5;
  int m_off = (w & 3) * 32, n_off = (w >> 2) * 64;
  int mr = lane & 15, kh = lane >> 4;
  v8f acc[2][4] = {};

  for (int kk = 0; kk < K; kk += 32) {
    { // stage A: 128x32, row-major, vectorized 16B
      int m = t >> 1, ks = (t & 1) * 16;
      const uint4* gp = (const uint4*)(A + (size_t)(m0 + m) * K + kk + ks);
      uint4 r0 = gp[0], r1 = gp[1];
      *(uint4*)&lA[m * 32 + ks] = r0;
      *(uint4*)&lA[m * 32 + ks + 8] = r1;
    }
    { // stage B: 32x128 global -> [n][k] LDS (transpose)
      int k = t >> 3, ns = (t & 7) * 16;
      const uint4* gp = (const uint4*)(B + (size_t)(kk + k) * N + n0 + ns);
      union { uint4 q[2]; unsigned short s[16]; } u;
      u.q[0] = gp[0]; u.q[1] = gp[1];
#pragma unroll
      for (int i = 0; i < 16; ++i) lB[(ns + i) * 32 + k] = u.s[i];
      if (kk + 32 < K)  // gfx1250 global_prefetch for next k-tile of B
        __builtin_prefetch(B + (size_t)(kk + 32 + k) * N + n0 + ns, 0, 1);
    }
    __syncthreads();

    BF16Frag a[2], b[4];
#pragma unroll
    for (int i = 0; i < 2; ++i) {
      int m = m_off + i * 16 + mr;
      a[i].u[0] = *(const uint4*)&lA[m * 32 + kh * 8];        // K = kh*8 .. +7
      a[i].u[1] = *(const uint4*)&lA[m * 32 + kh * 8 + 16];   // K = kh*8+16 .. +23
    }
#pragma unroll
    for (int j = 0; j < 4; ++j) {
      int n = n_off + j * 16 + mr;
      b[j].u[0] = *(const uint4*)&lB[n * 32 + kh * 8];        // K = kh*8 .. +7
      b[j].u[1] = *(const uint4*)&lB[n * 32 + kh * 8 + 16];   // K = kh*8+16 .. +23
    }
#pragma unroll
    for (int i = 0; i < 2; ++i)
#pragma unroll
      for (int j = 0; j < 4; ++j)
        acc[i][j] = __builtin_amdgcn_wmma_f32_16x16x32_bf16(
            false, a[i].v, false, b[j].v, (short)0, acc[i][j], false, false);
    __syncthreads();
  }

#pragma unroll
  for (int i = 0; i < 2; ++i)
#pragma unroll
    for (int j = 0; j < 4; ++j) {
      int gn = n0 + n_off + j * 16 + mr;
      float bv = bias[gn];
#pragma unroll
      for (int r = 0; r < 8; ++r) {
        int gm = m0 + m_off + i * 16 + kh * 8 + r;
        float v = acc[i][j][r] + bv;
        if (EPI == 0) {
          outbf[(size_t)gm * N + gn] = f2bf(v);
        } else {
          outf[(size_t)gm * N + gn] =
              xres[(size_t)gm * N + gn] + gate[gn] * v;
        }
      }
    }
}

// ---------------------------------------------------------------------------
// 5) RMSNorm + RoPE on q,k in-place in y (bf16). block = (l,h), 128 threads.
// ---------------------------------------------------------------------------
__global__ void k_rmsrope(unsigned short* __restrict__ y,
                          const float* __restrict__ qw, const float* __restrict__ kw,
                          const float* __restrict__ fcos, const float* __restrict__ fsin) {
  int l = blockIdx.x, h = blockIdx.y, d = threadIdx.x;
  __shared__ float red[128];
  size_t base = (size_t)l * NY + h * HD;
  float q  = bf2f(y[base + d]);
  float qp = bf2f(y[base + (d ^ 1)]);
  float k  = bf2f(y[base + HID + d]);
  float kp = bf2f(y[base + HID + (d ^ 1)]);

  red[d] = q * q; __syncthreads();
  for (int o = 64; o > 0; o >>= 1) { if (d < o) red[d] += red[d + o]; __syncthreads(); }
  float qr = rsqrtf(red[0] / HD + EPSV); __syncthreads();
  red[d] = k * k; __syncthreads();
  for (int o = 64; o > 0; o >>= 1) { if (d < o) red[d] += red[d + o]; __syncthreads(); }
  float kr = rsqrtf(red[0] / HD + EPSV); __syncthreads();

  float qn  = q * qr * qw[d],  qpn = qp * qr * qw[d ^ 1];
  float kn  = k * kr * kw[d],  kpn = kp * kr * kw[d ^ 1];
  float qo = qn, ko = kn;
  if (l < IMG) {
    float c = fcos[(size_t)l * HD + d], s = fsin[(size_t)l * HD + d];
    float qrot = (d & 1) ? qpn : -qpn;
    float krot = (d & 1) ? kpn : -kpn;
    qo = qn * c + qrot * s;
    ko = kn * c + krot * s;
  }
  __syncthreads();                       // all raw loads done before rewrite
  y[base + d] = f2bf(qo);
  y[base + HID + d] = f2bf(ko);
}

// ---------------------------------------------------------------------------
// 6) GELU(tanh) on mlp part of y -> concat cols [HID, HID+MLP)
// ---------------------------------------------------------------------------
__global__ void k_gelu(const unsigned short* __restrict__ y,
                       unsigned short* __restrict__ cat) {
  size_t idx = (size_t)blockIdx.x * 256 + threadIdx.x;
  if (idx >= (size_t)LQ * MLP) return;
  int l = (int)(idx / MLP), j = (int)(idx % MLP);
  float v = bf2f(y[(size_t)l * NY + 3 * HID + j]);
  float g = 0.5f * v * (1.f + tanhf(0.7978845608f * (v + 0.044715f * v * v * v)));
  cat[(size_t)l * NC + HID + j] = f2bf(g);
}

// ---------------------------------------------------------------------------
// 7) Flash attention. block = (head, q-tile of 128). 8 waves x 16 q-rows.
//    Online softmax with wave32 16-lane shfl_xor row reductions.
// ---------------------------------------------------------------------------
__global__ __launch_bounds__(256) void k_attn(const unsigned short* __restrict__ y,
                                              unsigned short* __restrict__ cat) {
  int h = blockIdx.x, qt = blockIdx.y;
  int t = threadIdx.x, lane = t & 31, w = t >> 5;
  int mr = lane & 15, kh = lane >> 4;
  __shared__ unsigned short lK[64 * 128];   // [kv][d]
  __shared__ unsigned short lV[128 * 64];   // [d][kv] (transposed)
  __shared__ unsigned short lP[8][16 * 64]; // per-wave P tile [m][kv]
  int q0 = qt * 128 + w * 16;

  BF16Frag qf[4];
  {
    size_t qb = (size_t)(q0 + mr) * NY + h * HD;
#pragma unroll
    for (int j = 0; j < 4; ++j) {
      qf[j].u[0] = *(const uint4*)&y[qb + j * 32 + kh * 8];
      qf[j].u[1] = *(const uint4*)&y[qb + j * 32 + kh * 8 + 16];
    }
  }
  v8f o[8] = {};
  float mrun[8], lrun[8];
#pragma unroll
  for (int r = 0; r < 8; ++r) { mrun[r] = -1e30f; lrun[r] = 0.f; }
  const float rsc = 0.08838834764831845f;   // 1/sqrt(HD)

  for (int kv0 = 0; kv0 < LQ; kv0 += 64) {
    { // stage K (row-major) and V (transposed)
      int kv = t >> 2, ds = (t & 3) * 32;
      const uint4* gk = (const uint4*)&y[(size_t)(kv0 + kv) * NY + HID + h * HD + ds];
      uint4 a0 = gk[0], a1 = gk[1], a2 = gk[2], a3 = gk[3];
      *(uint4*)&lK[kv * 128 + ds]      = a0;
      *(uint4*)&lK[kv * 128 + ds + 8]  = a1;
      *(uint4*)&lK[kv * 128 + ds + 16] = a2;
      *(uint4*)&lK[kv * 128 + ds + 24] = a3;
      union { uint4 q[4]; unsigned short s[32]; } uv;
      const uint4* gv = (const uint4*)&y[(size_t)(kv0 + kv) * NY + 2 * HID + h * HD + ds];
      uv.q[0] = gv[0]; uv.q[1] = gv[1]; uv.q[2] = gv[2]; uv.q[3] = gv[3];
#pragma unroll
      for (int i = 0; i < 32; ++i) lV[(ds + i) * 64 + kv] = uv.s[i];
    }
    __syncthreads();

    // S = (Q K^T) * rsc  (16 x 64 per wave)
    v8f s[4] = {};
#pragma unroll
    for (int j = 0; j < 4; ++j) {
      int n = j * 16 + mr;
#pragma unroll
      for (int kd = 0; kd < 4; ++kd) {
        BF16Frag b;
        b.u[0] = *(const uint4*)&lK[n * 128 + kd * 32 + kh * 8];
        b.u[1] = *(const uint4*)&lK[n * 128 + kd * 32 + kh * 8 + 16];
        s[j] = __builtin_amdgcn_wmma_f32_16x16x32_bf16(
            false, qf[kd].v, false, b.v, (short)0, s[j], false, false);
      }
    }

    // online softmax update per owned row r (row = kh*8+r, cols in 16-lane half)
#pragma unroll
    for (int r = 0; r < 8; ++r) {
      float mx = -1e30f;
#pragma unroll
      for (int j = 0; j < 4; ++j) mx = fmaxf(mx, s[j][r] * rsc);
#pragma unroll
      for (int xx = 1; xx < 16; xx <<= 1) mx = fmaxf(mx, __shfl_xor(mx, xx, 32));
      float mnew = fmaxf(mrun[r], mx);
      float fr = __expf(mrun[r] - mnew);
      mrun[r] = mnew;
#pragma unroll
      for (int dt = 0; dt < 8; ++dt) o[dt][r] *= fr;
      float rs = 0.f;
#pragma unroll
      for (int j = 0; j < 4; ++j) {
        float p = __expf(s[j][r] * rsc - mnew);
        rs += p;
        lP[w][(kh * 8 + r) * 64 + j * 16 + mr] = f2bf(p);
      }
#pragma unroll
      for (int xx = 1; xx < 16; xx <<= 1) rs += __shfl_xor(rs, xx, 32);
      lrun[r] = lrun[r] * fr + rs;
    }

    // O += P @ V   (A from per-wave lP, B from transposed lV)
#pragma unroll
    for (int dt = 0; dt < 8; ++dt) {
      int n = dt * 16 + mr;
#pragma unroll
      for (int k2 = 0; k2 < 2; ++k2) {
        BF16Frag a, b;
        a.u[0] = *(const uint4*)&lP[w][mr * 64 + k2 * 32 + kh * 8];
        a.u[1] = *(const uint4*)&lP[w][mr * 64 + k2 * 32 + kh * 8 + 16];
        b.u[0] = *(const uint4*)&lV[n * 64 + k2 * 32 + kh * 8];
        b.u[1] = *(const uint4*)&lV[n * 64 + k2 * 32 + kh * 8 + 16];
        o[dt] = __builtin_amdgcn_wmma_f32_16x16x32_bf16(
            false, a.v, false, b.v, (short)0, o[dt], false, false);
      }
    }
    __syncthreads();
  }

#pragma unroll
  for (int dt = 0; dt < 8; ++dt)
#pragma unroll
    for (int r = 0; r < 8; ++r) {
      int gm = q0 + kh * 8 + r;
      int gn = h * HD + dt * 16 + mr;
      cat[(size_t)gm * NC + gn] = f2bf(o[dt][r] / lrun[r]);
    }
}

// ---------------------------------------------------------------------------
// Launch
// ---------------------------------------------------------------------------
extern "C" void kernel_launch(void* const* d_in, const int* in_sizes, int n_in,
                              void* d_out, int out_size, void* d_ws, size_t ws_size,
                              hipStream_t stream) {
  const float* x     = (const float*)d_in[0];
  const float* vec   = (const float*)d_in[1];
  const float* w_mod = (const float*)d_in[2];
  const float* b_mod = (const float*)d_in[3];
  const float* w1    = (const float*)d_in[4];
  const float* b1    = (const float*)d_in[5];
  const float* w2    = (const float*)d_in[6];
  const float* b2    = (const float*)d_in[7];
  const float* qnw   = (const float*)d_in[8];
  const float* knw   = (const float*)d_in[9];
  const float* fcos  = (const float*)d_in[10];
  const float* fsin  = (const float*)d_in[11];
  // d_in[12] = txt_len (compile-time constant TXT here)

  char* ws = (char*)d_ws;
  float*          mod  = (float*)(ws + OFF_MOD);
  unsigned short* xmod = (unsigned short*)(ws + OFF_XMOD);
  unsigned short* w1b  = (unsigned short*)(ws + OFF_W1B);
  unsigned short* ybf  = (unsigned short*)(ws + OFF_Y);
  unsigned short* w2b  = (unsigned short*)(ws + OFF_W2B);
  unsigned short* cat  = (unsigned short*)(ws + OFF_CAT);
  const float* gate = mod + 2 * HID;

  k_mod<<<(3 * HID) / 256, 256, 0, stream>>>(vec, w_mod, b_mod, mod);

  size_t n1 = (size_t)HID * NY, n2 = (size_t)NC * HID;
  k_cvt<<<(unsigned)((n1 + 1023) / 1024), 256, 0, stream>>>(w1, w1b, n1);
  k_cvt<<<(unsigned)((n2 + 1023) / 1024), 256, 0, stream>>>(w2, w2b, n2);

  k_ln<<<LQ, 256, 0, stream>>>(x, mod, xmod);

  dim3 g1(NY / 128, LQ / 128);   // 168 x 16
  k_gemm<0><<<g1, 256, 0, stream>>>(xmod, w1b, b1, HID, NY, ybf,
                                    nullptr, nullptr, nullptr);

  k_rmsrope<<<dim3(LQ, NH), 128, 0, stream>>>(ybf, qnw, knw, fcos, fsin);

  k_gelu<<<(unsigned)(((size_t)LQ * MLP + 255) / 256), 256, 0, stream>>>(ybf, cat);

  k_attn<<<dim3(NH, LQ / 128), 256, 0, stream>>>(ybf, cat);

  dim3 g2(HID / 128, LQ / 128);  // 24 x 16
  k_gemm<1><<<g2, 256, 0, stream>>>(cat, w2b, b2, NC, HID, nullptr,
                                    x, gate, (float*)d_out);
}
